// AnatomicalGCN_30511447671339
// MI455X (gfx1250) — matrix operand order repaired and checked by reading, not compiled, not run on hardware
//
#include <hip/hip_runtime.h>
#include <hip/hip_bf16.h>

typedef __attribute__((ext_vector_type(16))) _Float16 v16h;
typedef __attribute__((ext_vector_type(8)))  float    v8f;

// ---------------------------------------------------------------------------
// WMMA helpers (CDNA5: D = A(16x32 f16) x B(32x16 f16) + C(16x16 f32))
// ---------------------------------------------------------------------------
__device__ __forceinline__ v8f wmma32(v16h a, v16h b, v8f c) {
    return __builtin_amdgcn_wmma_f32_16x16x32_f16(false, a, false, b, (short)0, c, false, false);
}

// A-fragment gather from an LDS fp32 tile (row-major).
// ISA layout: lane L holds row M=L%16; half e maps to K=(e<8?e:e+8)+8*(L/16),
// i.e. two runs of 8 consecutive K values -> 4 x ds_load_b128 + packed cvt.
// Requires: stride % 4 == 0, kbase % 32 == 0, 16B-aligned LDS base.
__device__ __forceinline__ v16h load_a_frag(const float* lds, int row_base, int stride,
                                            int kbase, int lane) {
    const int m    = lane & 15;
    const int koff = (lane >> 4) * 8;
    const float* p = lds + (row_base + m) * stride + kbase + koff;
    float4 a0 = *(const float4*)(p);
    float4 a1 = *(const float4*)(p + 4);
    float4 b0 = *(const float4*)(p + 16);
    float4 b1 = *(const float4*)(p + 20);
    v16h r;
    r[0]  = (_Float16)a0.x; r[1]  = (_Float16)a0.y; r[2]  = (_Float16)a0.z; r[3]  = (_Float16)a0.w;
    r[4]  = (_Float16)a1.x; r[5]  = (_Float16)a1.y; r[6]  = (_Float16)a1.z; r[7]  = (_Float16)a1.w;
    r[8]  = (_Float16)b0.x; r[9]  = (_Float16)b0.y; r[10] = (_Float16)b0.z; r[11] = (_Float16)b0.w;
    r[12] = (_Float16)b1.x; r[13] = (_Float16)b1.y; r[14] = (_Float16)b1.z; r[15] = (_Float16)b1.w;
    return r;
}

// D/C fp32 tile: lane L -> col N=L%16, vgpr r -> row M=r+8*(L/16)
__device__ __forceinline__ void store_d_lds(float* lds, v8f d, int row_base, int col_base,
                                            int stride, int lane) {
    const int n  = lane & 15;
    const int mo = (lane >> 4) * 8;
#pragma unroll
    for (int r = 0; r < 8; ++r)
        lds[(row_base + mo + r) * stride + col_base + n] = d[r];
}

// ---------------------------------------------------------------------------
// Weight pre-swizzle: fp32 KxN row-major -> f16 B-fragment layout.
// B-fragment: within a 32-K chunk, lane = (n%16) + 16*((k%16)>=8), e = k%8 + 8*(k%32>=16),
// stored so each lane reads a contiguous v16h (32B).
// ---------------------------------------------------------------------------
__device__ __forceinline__ void swizzleB(const float* W, _Float16* dst, int K, int N, int idx) {
    int k  = idx / N, n = idx % N;
    int kc = k >> 5, kr = k & 31;
    int nt = n >> 4, nr = n & 15;
    int lane = nr | ((kr & 8) << 1);
    int e    = (kr & 7) | ((kr & 16) >> 1);
    int KC   = K >> 5;
    dst[((size_t)((nt * KC + kc) * 32 + lane)) * 16 + e] = (_Float16)W[idx];
}

__global__ void prep_weights(const float* __restrict__ qkv_w, const float* __restrict__ out_w,
                             const float* __restrict__ ff1_w, const float* __restrict__ ff2_w,
                             _Float16* __restrict__ hw) {
    int t = blockIdx.x * blockDim.x + threadIdx.x;
    if (t < 2 * 12288) { int l = t / 12288, e = t % 12288;
        swizzleB(qkv_w + l * 12288, hw + l * 12288, 64, 192, e); return; }
    t -= 2 * 12288;
    if (t < 2 * 4096) { int l = t / 4096, e = t % 4096;
        swizzleB(out_w + l * 4096, hw + 24576 + l * 4096, 64, 64, e); return; }
    t -= 2 * 4096;
    if (t < 2 * 131072) { int l = t / 131072, e = t % 131072;
        swizzleB(ff1_w + l * 131072, hw + 32768 + l * 131072, 64, 2048, e); return; }
    t -= 2 * 131072;
    if (t < 2 * 131072) { int l = t / 131072, e = t % 131072;
        swizzleB(ff2_w + l * 131072, hw + 294912 + l * 131072, 2048, 64, e); return; }
}

// ---------------------------------------------------------------------------
// Region GCN: 2-layer GCN with normalized adjacency, mean-pool over nodes.
// One block = 64 threads (thread d owns output channel d), 16 frames/block.
// ---------------------------------------------------------------------------
template <int N, int REGION, int POS>
__global__ __launch_bounds__(64) void gcn_kernel(const float* __restrict__ xr,
                                                 const float* __restrict__ w1,
                                                 const float* __restrict__ b1,
                                                 const float* __restrict__ w2,
                                                 const float* __restrict__ b2,
                                                 float* __restrict__ X) {
    __shared__ float A_sh[20 * 20];
    __shared__ float dinv[20];
    __shared__ float xsh[40];
    __shared__ float h1sh[20 * 65];

    const int d = threadIdx.x;
    if (d == 0) {
        for (int i = 0; i < N; ++i)
            for (int j = 0; j < N; ++j) A_sh[i * 20 + j] = 0.f;
        for (int i = 0; i + 1 < N; ++i) { A_sh[i * 20 + i + 1] = 1.f; A_sh[(i + 1) * 20 + i] = 1.f; }
        for (int i = 0; i + 2 < N; ++i) { A_sh[i * 20 + i + 2] = 1.f; A_sh[(i + 2) * 20 + i] = 1.f; }
        for (int i = 0; i < N; ++i) A_sh[i * 20 + i] += 1.f;
        for (int i = 0; i < N; ++i) {
            float s = 0.f;
            for (int j = 0; j < N; ++j) s += A_sh[i * 20 + j];
            dinv[i] = rsqrtf(s);
        }
        for (int i = 0; i < N; ++i)
            for (int j = 0; j < N; ++j) A_sh[i * 20 + j] *= dinv[i] * dinv[j];
    }
    const float w1a = w1[REGION * 128 + d];
    const float w1b = w1[REGION * 128 + 64 + d];
    const float b1v = b1[REGION * 64 + d];
    const float b2v = b2[REGION * 64 + d];
    float w2col[64];
#pragma unroll
    for (int k = 0; k < 64; ++k) w2col[k] = w2[REGION * 4096 + k * 64 + d];
    __syncthreads();

    for (int f = 0; f < 16; ++f) {
        const int bt = blockIdx.x * 16 + f;
        for (int i = d; i < N * 2; i += 64) xsh[i] = xr[(size_t)bt * N * 2 + i];
        __syncthreads();
        float xw[N];
#pragma unroll
        for (int n = 0; n < N; ++n) xw[n] = xsh[n * 2] * w1a + xsh[n * 2 + 1] * w1b;
#pragma unroll
        for (int m = 0; m < N; ++m) {
            float s = b1v;
#pragma unroll
            for (int n = 0; n < N; ++n) s += A_sh[m * 20 + n] * xw[n];
            h1sh[m * 65 + d] = fmaxf(s, 0.f);
        }
        __syncthreads();
        float g[N];
#pragma unroll
        for (int n = 0; n < N; ++n) {
            float s = 0.f;
#pragma unroll
            for (int k = 0; k < 64; ++k) s += h1sh[n * 65 + k] * w2col[k];
            g[n] = s;
        }
        float accm = 0.f;
#pragma unroll
        for (int m = 0; m < N; ++m) {
            float s = b2v;
#pragma unroll
            for (int n = 0; n < N; ++n) s += A_sh[m * 20 + n] * g[n];
            accm += fmaxf(s, 0.f);
        }
        X[((size_t)bt * 6 + POS) * 64 + d] = accm * (1.f / N);
        __syncthreads();
    }
}

// ---------------------------------------------------------------------------
// Fused MHA block: QKV (WMMA) -> softmax attention (S=6, VALU) -> out proj
// (WMMA) -> residual + LN1.  8 samples (48 token rows) per block, 8 waves.
// Strides: xs/ob = 68 (=4 mod 64 -> conflict-free 16-lane b128), qk = 197.
// ---------------------------------------------------------------------------
#define XS 68
#define QS 197
__global__ __launch_bounds__(256) void attn_kernel(const float* __restrict__ Xin,
                                                   float* __restrict__ Xout,
                                                   const _Float16* __restrict__ swqkv,
                                                   const float* __restrict__ qkv_b,
                                                   const _Float16* __restrict__ swout,
                                                   const float* __restrict__ out_b,
                                                   const float* __restrict__ ln_g,
                                                   const float* __restrict__ ln_b) {
    __shared__ __align__(16) float xs[48 * XS];
    __shared__ __align__(16) float qk[48 * QS];
    __shared__ __align__(16) float ob[48 * XS];
    const int tid = threadIdx.x, lane = tid & 31, wave = tid >> 5;
    const int row0 = blockIdx.x * 48;

    // vectorized tile load: 48 rows x 16 float4
    for (int i = tid; i < 48 * 16; i += 256) {
        float4 v = ((const float4*)(Xin + (size_t)row0 * 64))[i];
        *(float4*)(xs + (i >> 4) * XS + (i & 15) * 4) = v;
    }
    __syncthreads();

    // QKV: M=48 (3 tiles) x N=192 (12 tiles), K=64
    for (int t = wave; t < 36; t += 8) {
        const int mt = t / 12, nt = t % 12;
        const float bias = qkv_b[nt * 16 + (lane & 15)];
        v8f acc = {bias, bias, bias, bias, bias, bias, bias, bias};
#pragma unroll
        for (int kc = 0; kc < 2; ++kc) {
            v16h af = load_a_frag(xs, mt * 16, XS, kc * 32, lane);
            v16h bf = *(const v16h*)(swqkv + (size_t)((nt * 2 + kc) * 32 + lane) * 16);
            acc = wmma32(af, bf, acc);
        }
        store_d_lds(qk, acc, mt * 16, nt * 16, QS, lane);
    }
    __syncthreads();

    // attention: 16 threads per sample -> (head, query-row) pairs
    {
        const int sl = tid >> 4;
        if (sl < 8) {
            const int u = tid & 15, h = u & 3, qs0 = u >> 2;
            const int rb = sl * 6;
            for (int q = qs0; q < 6; q += 4) {
                float sc[6], mx = -1e30f;
#pragma unroll
                for (int t2 = 0; t2 < 6; ++t2) {
                    float dd = 0.f;
#pragma unroll
                    for (int j = 0; j < 16; ++j)
                        dd += qk[(rb + q) * QS + h * 16 + j] * qk[(rb + t2) * QS + 64 + h * 16 + j];
                    sc[t2] = dd * 0.25f;
                    mx = fmaxf(mx, sc[t2]);
                }
                float se = 0.f;
#pragma unroll
                for (int t2 = 0; t2 < 6; ++t2) { sc[t2] = __expf(sc[t2] - mx); se += sc[t2]; }
                const float inv = 1.f / se;
#pragma unroll
                for (int j = 0; j < 16; ++j) {
                    float o = 0.f;
#pragma unroll
                    for (int t2 = 0; t2 < 6; ++t2)
                        o += sc[t2] * qk[(rb + t2) * QS + 128 + h * 16 + j];
                    ob[(rb + q) * XS + h * 16 + j] = o * inv;
                }
            }
        }
    }
    __syncthreads();

    // out projection: M=48 (3 tiles) x N=64 (4 tiles), K=64 -> into qk[:,0:64]
    for (int t = wave; t < 12; t += 8) {
        const int mt = t >> 2, nt = t & 3;
        const float bias = out_b[nt * 16 + (lane & 15)];
        v8f acc = {bias, bias, bias, bias, bias, bias, bias, bias};
#pragma unroll
        for (int kc = 0; kc < 2; ++kc) {
            v16h af = load_a_frag(ob, mt * 16, XS, kc * 32, lane);
            v16h bf = *(const v16h*)(swout + (size_t)((nt * 2 + kc) * 32 + lane) * 16);
            acc = wmma32(af, bf, acc);
        }
        store_d_lds(qk, acc, mt * 16, nt * 16, QS, lane);
    }
    __syncthreads();

    // residual + LN1
    for (int r = tid; r < 48; r += 256) {
        float mu = 0.f;
        for (int c = 0; c < 64; ++c) mu += xs[r * XS + c] + qk[r * QS + c];
        mu *= (1.f / 64.f);
        float var = 0.f;
        for (int c = 0; c < 64; ++c) {
            float dv = xs[r * XS + c] + qk[r * QS + c] - mu;
            var += dv * dv;
        }
        var *= (1.f / 64.f);
        const float rs = rsqrtf(var + 1e-5f);
        for (int c = 0; c < 64; ++c)
            Xout[(size_t)(row0 + r) * 64 + c] =
                (xs[r * XS + c] + qk[r * QS + c] - mu) * rs * ln_g[c] + ln_b[c];
    }
}

// ---------------------------------------------------------------------------
// Fused FFN: relu(x@W1+b1)@W2+b2 + residual + LN2. 16 rows/block, 8 waves.
// FF intermediate streamed through LDS in 256-col chunks (never hits HBM).
// Split-K: waves (nt,kh) -> 4 output tiles x 2 K-halves, combined via LDS.
// H stride 260 (=4 mod 64): conflict-free b128 fragment gathers.
// ---------------------------------------------------------------------------
#define HS 260
__global__ __launch_bounds__(256) void ffn_kernel(const float* __restrict__ Xin,
                                                  float* __restrict__ Xout,
                                                  const _Float16* __restrict__ swff1,
                                                  const float* __restrict__ ff1_b,
                                                  const _Float16* __restrict__ swff2,
                                                  const float* __restrict__ ff2_b,
                                                  const float* __restrict__ ln_g,
                                                  const float* __restrict__ ln_b) {
    __shared__ __align__(16) float xt[16 * XS];
    __shared__ __align__(16) float H[16 * HS];
    __shared__ __align__(16) float rbuf[16 * XS];
    const int tid = threadIdx.x, lane = tid & 31, wave = tid >> 5;
    const int row0 = blockIdx.x * 16;

    for (int i = tid; i < 16 * 16; i += 256) {
        float4 v = ((const float4*)(Xin + (size_t)row0 * 64))[i];
        *(float4*)(xt + (i >> 4) * XS + (i & 15) * 4) = v;
    }
    __syncthreads();

    const int nt = wave & 3, kh = wave >> 2;
    v8f acc = {};

    // x A-fragments are chunk-invariant: hoist
    v16h xa0 = load_a_frag(xt, 0, XS, 0, lane);
    v16h xa1 = load_a_frag(xt, 0, XS, 32, lane);

    for (int c = 0; c < 8; ++c) {
        if (c < 7) {
            __builtin_prefetch(swff1 + (size_t)((((c + 1) * 16 + wave) * 2) * 32 + lane) * 16, 0, 1);
            __builtin_prefetch(swff2 + (size_t)((nt * 64 + (c + 1) * 8 + kh * 4) * 32 + lane) * 16, 0, 1);
        }
        // FF1 chunk: H[16 x 256] = relu(x @ W1[:, c*256 : c*256+256] + b1)
        for (int t = wave; t < 16; t += 8) {
            const int ncol = c * 256 + t * 16 + (lane & 15);
            const float bias = ff1_b[ncol];
            v8f a1 = {bias, bias, bias, bias, bias, bias, bias, bias};
            v16h b0  = *(const v16h*)(swff1 + (size_t)(((c * 16 + t) * 2 + 0) * 32 + lane) * 16);
            v16h b1v = *(const v16h*)(swff1 + (size_t)(((c * 16 + t) * 2 + 1) * 32 + lane) * 16);
            a1 = wmma32(xa0, b0, a1);
            a1 = wmma32(xa1, b1v, a1);
            const int mo = (lane >> 4) * 8, nn = t * 16 + (lane & 15);
#pragma unroll
            for (int r = 0; r < 8; ++r) H[(mo + r) * HS + nn] = fmaxf(a1[r], 0.f);
        }
        __syncthreads();
        // FF2 partial: acc += H[:, kh*128:+128] @ W2 chunk
#pragma unroll
        for (int kk = 0; kk < 4; ++kk) {
            v16h af = load_a_frag(H, 0, HS, kh * 128 + kk * 32, lane);
            const int kcg = c * 8 + kh * 4 + kk;
            v16h bf = *(const v16h*)(swff2 + (size_t)((nt * 64 + kcg) * 32 + lane) * 16);
            acc = wmma32(af, bf, acc);
        }
        __syncthreads();
    }

    // combine split-K halves, add bias + residual
    if (kh == 1) {
        const int mo = (lane >> 4) * 8, nn = nt * 16 + (lane & 15);
#pragma unroll
        for (int r = 0; r < 8; ++r) rbuf[(mo + r) * XS + nn] = acc[r];
    }
    __syncthreads();
    if (kh == 0) {
        const int mo = (lane >> 4) * 8, nn = nt * 16 + (lane & 15);
#pragma unroll
        for (int r = 0; r < 8; ++r) {
            const int m = mo + r;
            H[m * XS + nn] = acc[r] + rbuf[m * XS + nn] + ff2_b[nn] + xt[m * XS + nn];
        }
    }
    __syncthreads();
    // LN2 + store
    if (tid < 16) {
        const int m = tid;
        float mu = 0.f;
        for (int c = 0; c < 64; ++c) mu += H[m * XS + c];
        mu *= (1.f / 64.f);
        float var = 0.f;
        for (int c = 0; c < 64; ++c) { float dv = H[m * XS + c] - mu; var += dv * dv; }
        var *= (1.f / 64.f);
        const float rs = rsqrtf(var + 1e-5f);
        for (int c = 0; c < 64; ++c)
            Xout[(size_t)(row0 + m) * 64 + c] = (H[m * XS + c] - mu) * rs * ln_g[c] + ln_b[c];
    }
}

// ---------------------------------------------------------------------------
// mean over 6 tokens and T frames -> classifier (64->32->2)
// ---------------------------------------------------------------------------
__global__ __launch_bounds__(64) void reduce_cls(const float* __restrict__ X,
                                                 const float* __restrict__ w1,
                                                 const float* __restrict__ b1,
                                                 const float* __restrict__ w2,
                                                 const float* __restrict__ b2,
                                                 float* __restrict__ out) {
    __shared__ float vrow[64];
    __shared__ float hbuf[32];
    const int b = blockIdx.x, d = threadIdx.x;
    const float* p = X + (size_t)b * 512 * 6 * 64;
    float s = 0.f;
    for (int i = 0; i < 3072; ++i) s += p[(size_t)i * 64 + d];
    vrow[d] = s * (1.f / 3072.f);
    __syncthreads();
    if (d < 32) {
        float h = b1[d];
        for (int k = 0; k < 64; ++k) h += vrow[k] * w1[k * 32 + d];
        hbuf[d] = fmaxf(h, 0.f);
    }
    __syncthreads();
    if (d < 2) {
        float o = b2[d];
        for (int k = 0; k < 32; ++k) o += hbuf[k] * w2[k * 2 + d];
        out[b * 2 + d] = o;
    }
}

// ---------------------------------------------------------------------------
extern "C" void kernel_launch(void* const* d_in, const int* in_sizes, int n_in,
                              void* d_out, int out_size, void* d_ws, size_t ws_size,
                              hipStream_t stream) {
    const float* mouth  = (const float*)d_in[0];
    const float* nose   = (const float*)d_in[1];
    const float* leye   = (const float*)d_in[2];
    const float* reye   = (const float*)d_in[3];
    const float* ljaw   = (const float*)d_in[4];
    const float* rjaw   = (const float*)d_in[5];
    const float* gcn_w1 = (const float*)d_in[6];
    const float* gcn_b1 = (const float*)d_in[7];
    const float* gcn_w2 = (const float*)d_in[8];
    const float* gcn_b2 = (const float*)d_in[9];
    const float* qkv_w  = (const float*)d_in[10];
    const float* qkv_b  = (const float*)d_in[11];
    const float* out_w  = (const float*)d_in[12];
    const float* out_b  = (const float*)d_in[13];
    const float* ff1_w  = (const float*)d_in[14];
    const float* ff1_b  = (const float*)d_in[15];
    const float* ff2_w  = (const float*)d_in[16];
    const float* ff2_b  = (const float*)d_in[17];
    const float* ln1_g  = (const float*)d_in[18];
    const float* ln1_b  = (const float*)d_in[19];
    const float* ln2_g  = (const float*)d_in[20];
    const float* ln2_b  = (const float*)d_in[21];
    const float* cls_w1 = (const float*)d_in[22];
    const float* cls_b1 = (const float*)d_in[23];
    const float* cls_w2 = (const float*)d_in[24];
    const float* cls_b2 = (const float*)d_in[25];
    (void)in_sizes; (void)n_in; (void)out_size; (void)ws_size;

    // workspace layout
    float* X0 = (float*)d_ws;                 // 49152 x 64
    float* X1 = X0 + 3145728;                 // 49152 x 64
    _Float16* hw    = (_Float16*)(X1 + 3145728);
    _Float16* swqkv = hw;                     // 2 x 12288
    _Float16* swout = hw + 24576;             // 2 x 4096
    _Float16* swff1 = hw + 32768;             // 2 x 131072
    _Float16* swff2 = hw + 294912;            // 2 x 131072

    prep_weights<<<(557056 + 255) / 256, 256, 0, stream>>>(qkv_w, out_w, ff1_w, ff2_w, hw);

    gcn_kernel<20, 0, 5><<<512, 64, 0, stream>>>(mouth, gcn_w1, gcn_b1, gcn_w2, gcn_b2, X0);
    gcn_kernel< 9, 1, 4><<<512, 64, 0, stream>>>(nose,  gcn_w1, gcn_b1, gcn_w2, gcn_b2, X0);
    gcn_kernel<11, 2, 2><<<512, 64, 0, stream>>>(leye,  gcn_w1, gcn_b1, gcn_w2, gcn_b2, X0);
    gcn_kernel<11, 3, 3><<<512, 64, 0, stream>>>(reye,  gcn_w1, gcn_b1, gcn_w2, gcn_b2, X0);
    gcn_kernel< 9, 4, 0><<<512, 64, 0, stream>>>(ljaw,  gcn_w1, gcn_b1, gcn_w2, gcn_b2, X0);
    gcn_kernel< 8, 5, 1><<<512, 64, 0, stream>>>(rjaw,  gcn_w1, gcn_b1, gcn_w2, gcn_b2, X0);

    for (int l = 0; l < 2; ++l) {
        attn_kernel<<<1024, 256, 0, stream>>>(X0, X1,
                                              swqkv + (size_t)l * 12288, qkv_b + l * 192,
                                              swout + (size_t)l * 4096,  out_b + l * 64,
                                              ln1_g + l * 64, ln1_b + l * 64);
        ffn_kernel<<<3072, 256, 0, stream>>>(X1, X0,
                                             swff1 + (size_t)l * 131072, ff1_b + l * 2048,
                                             swff2 + (size_t)l * 131072, ff2_b + l * 64,
                                             ln2_g + l * 64, ln2_b + l * 64);
    }

    reduce_cls<<<16, 64, 0, stream>>>(X0, cls_w1, cls_b1, cls_w2, cls_b2, (float*)d_out);
}
